// SparseAttention_14774687498830
// MI455X (gfx1250) — matrix operand order repaired and checked by reading, not compiled, not run on hardware
//
#include <hip/hip_runtime.h>
#include <hip/hip_bf16.h>

typedef __attribute__((ext_vector_type(16))) _Float16    v16h;
typedef __attribute__((ext_vector_type(8)))  float       v8f;
typedef __attribute__((ext_vector_type(8)))  unsigned int v8u;

using u16 = unsigned short;

// ---------------------------------------------------------------- helpers ---

__device__ __forceinline__ u16 f2h(float f) {
  _Float16 h = (_Float16)f;
  return __builtin_bit_cast(u16, h);
}

__device__ __forceinline__ v8f vzero8() {
  v8f z = {0.f, 0.f, 0.f, 0.f, 0.f, 0.f, 0.f, 0.f};
  return z;
}

__device__ __forceinline__ v8f wmma_f16(v16h a, v16h b, v8f c) {
  // D = A(16x32 f16) * B(32x16 f16) + C(16x16 f32)
  return __builtin_amdgcn_wmma_f32_16x16x32_f16(
      false, a, false, b, (short)0, c, false, false);
}

// A-fragment packed-pair K offset for VGPR r, lane-half h (16-bit A 16x32)
__device__ __forceinline__ int a_k0(int r, int h) {
  return ((r < 4) ? (2 * r) : (16 + 2 * (r - 4))) + 8 * h;
}

// gfx1250 async copy: 16B global -> LDS, tracked by ASYNCcnt (no VGPR staging).
// LDS aperture keeps the byte offset in addr[31:0], so truncation is the offset.
__device__ __forceinline__ void async_copy_b128(void* lds_dst, const void* gsrc) {
  unsigned lds = (unsigned)(size_t)lds_dst;
  unsigned long long ga = (unsigned long long)(size_t)gsrc;
  asm volatile("global_load_async_to_lds_b128 %0, %1, off"
               :: "v"(lds), "v"(ga)
               : "memory");
}

__device__ __forceinline__ void wait_async0() {
  asm volatile("s_wait_asynccnt 0" ::: "memory");
}

// ------------------------------------------------------------ fp32 -> fp16 ---

__global__ __launch_bounds__(256) void cvt_f32_f16_kernel(
    const float* __restrict__ in, u16* __restrict__ out, int n) {
  int i = blockIdx.x * blockDim.x + threadIdx.x;
  if (i < n) out[i] = f2h(in[i]);
}

// fp32 [K=1024][N=1024] -> fp16 transposed [N][K] (one-time, so uncoalesced
// stores are irrelevant; this buys row-major async tile copies in the GEMMs).
__global__ __launch_bounds__(256) void cvt_transpose_kernel(
    const float* __restrict__ in, u16* __restrict__ out) {
  int idx = blockIdx.x * 256 + threadIdx.x;  // 1M threads
  int k = idx >> 10, n = idx & 1023;
  out[(size_t)n * 1024 + k] = f2h(in[idx]);
}

// ------------------------------------------------- 64x64 GEMM tile (K=1024) ---
// A: [4096][1024] f16 row-major. Wt: [1024(N)][1024(K)] f16 row-major.
// Block = 128 threads = 4 waves; wave w owns rows [m0+16w, m0+16w+16).
// Double-buffered LDS, filled with async b128 copies overlapped with WMMA.

__device__ __forceinline__ void gemm_tile_64x64(
    const u16* __restrict__ A, const u16* __restrict__ Wt, int m0, int n0,
    u16 (&Alds)[2][64][32], u16 (&Blds)[2][64][32], v8f (&acc)[4]) {
  const int tid  = threadIdx.x;
  const int wave = tid >> 5;
  const int lane = tid & 31;
  const int Nl   = lane & 15;
  const int half = lane >> 4;
  const int mrow = wave * 16;

  auto issue = [&](int buf, int kc) {
    const int k0 = kc * 32;
#pragma unroll
    for (int i = 0; i < 2; ++i) {
      int flat4 = i * 128 + tid;
      int row = flat4 >> 2;  // 4 x b128 per 32-elem row
      int c8  = flat4 & 3;
      async_copy_b128(&Alds[buf][row][c8 * 8],
                      A + (size_t)(m0 + row) * 1024 + k0 + c8 * 8);
      async_copy_b128(&Blds[buf][row][c8 * 8],
                      Wt + (size_t)(n0 + row) * 1024 + k0 + c8 * 8);
    }
  };

  issue(0, 0);
  wait_async0();
  __syncthreads();

  for (int kc = 0; kc < 32; ++kc) {  // 1024 / 32
    const int cur = kc & 1;
    if (kc + 1 < 32) issue(cur ^ 1, kc + 1);  // prefetch next tile into other buffer

    v8u au;
#pragma unroll
    for (int r = 0; r < 8; ++r)
      au[r] = *reinterpret_cast<const unsigned int*>(&Alds[cur][mrow + Nl][a_k0(r, half)]);
    v16h af = __builtin_bit_cast(v16h, au);

#pragma unroll
    for (int s = 0; s < 4; ++s) {
      v8u bu;
#pragma unroll
      for (int r = 0; r < 8; ++r)
        bu[r] = *reinterpret_cast<const unsigned int*>(&Blds[cur][s * 16 + Nl][a_k0(r, half)]);
      v16h bf = __builtin_bit_cast(v16h, bu);
      acc[s] = wmma_f16(af, bf, acc[s]);
    }

    wait_async0();     // next tile's async copies done (this wave)
    __syncthreads();   // all waves done reading cur before it is overwritten
  }
}

// ------------------------------------------------------- QKV projection ------
// grid = (64 Mtiles, 16 Ntiles, 3 {q,k,v}).
// Q,K stored [B*H][T][64]; V stored transposed [B*H][64][T] for async V tiles.

__global__ __launch_bounds__(128) void qkv_gemm_kernel(
    const u16* __restrict__ xb,
    const u16* __restrict__ wqt, const u16* __restrict__ wkt, const u16* __restrict__ wvt,
    const float* __restrict__ bq, const float* __restrict__ bk, const float* __restrict__ bv,
    u16* __restrict__ Qo, u16* __restrict__ Ko, u16* __restrict__ Vo) {
  __shared__ u16 Alds[2][64][32];
  __shared__ u16 Blds[2][64][32];
  const int m0 = blockIdx.x * 64;
  const int n0 = blockIdx.y * 64;
  const int which = blockIdx.z;
  const u16*   Wt   = (which == 0) ? wqt : (which == 1) ? wkt : wvt;
  const float* bias = (which == 0) ? bq  : (which == 1) ? bk  : bv;
  u16*         out  = (which == 0) ? Qo  : (which == 1) ? Ko  : Vo;

  v8f acc[4];
#pragma unroll
  for (int s = 0; s < 4; ++s) acc[s] = vzero8();

  gemm_tile_64x64(xb, Wt, m0, n0, Alds, Blds, acc);

  const int tid = threadIdx.x, wave = tid >> 5, lane = tid & 31;
  const int Nl = lane & 15, half = lane >> 4;
#pragma unroll
  for (int s = 0; s < 4; ++s) {
    int n = n0 + s * 16 + Nl;
    int h = n >> 6, d = n & 63;
    float bsv = bias[n];
#pragma unroll
    for (int r = 0; r < 8; ++r) {
      int m = m0 + wave * 16 + r + 8 * half;  // global token row
      int b = m >> 11, t = m & 2047;
      u16 val = f2h(acc[s][r] + bsv);
      if (which == 2)
        out[((size_t)(b * 16 + h) * 64 + d) * 2048 + t] = val;    // V transposed
      else
        out[((size_t)(b * 16 + h) * 2048 + t) * 64 + d] = val;    // Q, K
    }
  }
}

// ------------------------------------------------------- flash attention -----
// grid = (32 qtiles, 32 b*h). Block = 128 = 4 waves; wave owns 16 query rows.
// K/V tiles double-buffered in LDS via async b128 copies.

__global__ __launch_bounds__(128) void flash_attn_kernel(
    const u16* __restrict__ Q, const u16* __restrict__ K, const u16* __restrict__ Vt,
    u16* __restrict__ O) {
  const int qt  = blockIdx.x;
  const int bh  = blockIdx.y;
  const int b   = bh >> 4, h = bh & 15;
  const int tid = threadIdx.x;
  const int wave = tid >> 5, lane = tid & 31;
  const int Nl = lane & 15, half = lane >> 4;
  const int q0 = qt * 64 + wave * 16;

  __shared__ u16 Klds[2][64][64];    // [key][dim]
  __shared__ u16 Vlds[2][64][64];    // [dim][key]
  __shared__ u16 Plds[4][16][64];    // per-wave P relayout scratch

  const u16* Qh  = Q  + (size_t)bh * 2048 * 64;
  const u16* Kh  = K  + (size_t)bh * 2048 * 64;
  const u16* Vth = Vt + (size_t)bh * 64 * 2048;

  // Preload Q fragments (A-layout), K-dim 64 = two 32-chunks, kept in regs
  v16h qf[2];
#pragma unroll
  for (int kc = 0; kc < 2; ++kc) {
    v8u tmp;
#pragma unroll
    for (int r = 0; r < 8; ++r) {
      int K0 = kc * 32 + a_k0(r, half);
      tmp[r] = *reinterpret_cast<const unsigned int*>(Qh + (size_t)(q0 + Nl) * 64 + K0);
    }
    qf[kc] = __builtin_bit_cast(v16h, tmp);
  }

  auto issue_kv = [&](int buf, int kt) {
    const int kb = kt * 64;
#pragma unroll
    for (int i = 0; i < 4; ++i) {
      int flat4 = i * 128 + tid;
      int row = flat4 >> 3;  // 8 x b128 per 64-elem row
      int c8  = flat4 & 7;
      async_copy_b128(&Klds[buf][row][c8 * 8],
                      Kh + (size_t)(kb + row) * 64 + c8 * 8);
      async_copy_b128(&Vlds[buf][row][c8 * 8],
                      Vth + (size_t)row * 2048 + kb + c8 * 8);
    }
  };

  float mrow[8], lrow[8];
  v8f oacc[4];
#pragma unroll
  for (int r = 0; r < 8; ++r) { mrow[r] = -1e30f; lrow[r] = 0.f; }
#pragma unroll
  for (int s = 0; s < 4; ++s) oacc[s] = vzero8();

  const int ktiles = qt + 1;  // causal: only key tiles at or before the diagonal
  issue_kv(0, 0);
  wait_async0();
  __syncthreads();

  for (int kt = 0; kt < ktiles; ++kt) {
    const int kb  = kt * 64;
    const int cur = kt & 1;
    if (kt + 1 < ktiles) issue_kv(cur ^ 1, kt + 1);

    // S = Q @ K^T : 4 N-subtiles x 2 K-chunks = 8 WMMA
    v8f sf[4];
#pragma unroll
    for (int s = 0; s < 4; ++s) sf[s] = vzero8();
#pragma unroll
    for (int kc = 0; kc < 2; ++kc) {
#pragma unroll
      for (int s = 0; s < 4; ++s) {
        v8u bu;
#pragma unroll
        for (int r = 0; r < 8; ++r) {
          int kk = kc * 32 + a_k0(r, half);
          bu[r] = *reinterpret_cast<const unsigned int*>(&Klds[cur][s * 16 + Nl][kk]);
        }
        v16h bf = __builtin_bit_cast(v16h, bu);
        sf[s] = wmma_f16(qf[kc], bf, sf[s]);
      }
    }

    // scale + causal mask (mask can only trigger on the diagonal tile)
    const bool diag = (kt == qt);
#pragma unroll
    for (int s = 0; s < 4; ++s) {
#pragma unroll
      for (int r = 0; r < 8; ++r) {
        float v = sf[s][r] * 0.125f;  // 1/sqrt(64)
        if (diag) {
          int qg = q0 + r + 8 * half;
          int kg = kb + s * 16 + Nl;
          if (kg > qg) v = -1e30f;
        }
        sf[s][r] = v;
      }
    }

    // per-row chunk max (reduce over 4 subtiles, then 16 lanes of the half)
    float cmax[8];
#pragma unroll
    for (int r = 0; r < 8; ++r) {
      float v = sf[0][r];
      v = fmaxf(v, sf[1][r]); v = fmaxf(v, sf[2][r]); v = fmaxf(v, sf[3][r]);
      cmax[r] = v;
    }
#pragma unroll
    for (int msk = 1; msk < 16; msk <<= 1) {
#pragma unroll
      for (int r = 0; r < 8; ++r) cmax[r] = fmaxf(cmax[r], __shfl_xor(cmax[r], msk, 32));
    }

    float alpha[8];
#pragma unroll
    for (int r = 0; r < 8; ++r) {
      float mn = fmaxf(mrow[r], cmax[r]);
      alpha[r] = __expf(mrow[r] - mn);
      mrow[r] = mn;
    }

    // P = exp(S - m); row sums
    float rsum[8];
#pragma unroll
    for (int r = 0; r < 8; ++r) rsum[r] = 0.f;
#pragma unroll
    for (int s = 0; s < 4; ++s) {
#pragma unroll
      for (int r = 0; r < 8; ++r) {
        float p = __expf(sf[s][r] - mrow[r]);
        sf[s][r] = p;
        rsum[r] += p;
      }
    }
#pragma unroll
    for (int msk = 1; msk < 16; msk <<= 1) {
#pragma unroll
      for (int r = 0; r < 8; ++r) rsum[r] += __shfl_xor(rsum[r], msk, 32);
    }
#pragma unroll
    for (int r = 0; r < 8; ++r) lrow[r] = lrow[r] * alpha[r] + rsum[r];

    // rescale running O
#pragma unroll
    for (int s = 0; s < 4; ++s)
#pragma unroll
      for (int r = 0; r < 8; ++r) oacc[s][r] *= alpha[r];

    // relayout P (C-layout -> A-layout) through per-wave LDS scratch
#pragma unroll
    for (int s = 0; s < 4; ++s)
#pragma unroll
      for (int r = 0; r < 8; ++r)
        Plds[wave][r + 8 * half][s * 16 + Nl] = f2h(sf[s][r]);

    // O += P @ V : 2 K-chunks x 4 dim-subtiles = 8 WMMA
#pragma unroll
    for (int kc = 0; kc < 2; ++kc) {
      v8u pu;
#pragma unroll
      for (int r = 0; r < 8; ++r) {
        int K0 = kc * 32 + a_k0(r, half);
        pu[r] = *reinterpret_cast<const unsigned int*>(&Plds[wave][Nl][K0]);
      }
      v16h pf = __builtin_bit_cast(v16h, pu);
#pragma unroll
      for (int s = 0; s < 4; ++s) {
        v8u vu;
#pragma unroll
        for (int r = 0; r < 8; ++r) {
          int kk = kc * 32 + a_k0(r, half);
          vu[r] = *reinterpret_cast<const unsigned int*>(&Vlds[cur][s * 16 + Nl][kk]);
        }
        v16h vf = __builtin_bit_cast(v16h, vu);
        oacc[s] = wmma_f16(pf, vf, oacc[s]);
      }
    }

    wait_async0();     // next K/V tile landed (this wave's copies)
    __syncthreads();   // all waves done with cur before overwrite
  }

  // epilogue: normalize, store f16 O in [B][T][H*64]
#pragma unroll
  for (int s = 0; s < 4; ++s) {
    int d = s * 16 + Nl;
#pragma unroll
    for (int r = 0; r < 8; ++r) {
      int t = q0 + r + 8 * half;
      float ov = oacc[s][r] / lrow[r];
      O[(size_t)(b * 2048 + t) * 1024 + h * 64 + d] = f2h(ov);
    }
  }
}

// ------------------------------------------------------- output projection ---

__global__ __launch_bounds__(128) void out_proj_kernel(
    const u16* __restrict__ Ob, const u16* __restrict__ wot,
    const float* __restrict__ bo, float* __restrict__ out) {
  __shared__ u16 Alds[2][64][32];
  __shared__ u16 Blds[2][64][32];
  const int m0 = blockIdx.x * 64;
  const int n0 = blockIdx.y * 64;

  v8f acc[4];
#pragma unroll
  for (int s = 0; s < 4; ++s) acc[s] = vzero8();

  gemm_tile_64x64(Ob, wot, m0, n0, Alds, Blds, acc);

  const int tid = threadIdx.x, wave = tid >> 5, lane = tid & 31;
  const int Nl = lane & 15, half = lane >> 4;
#pragma unroll
  for (int s = 0; s < 4; ++s) {
    int n = n0 + s * 16 + Nl;
    float bsv = bo[n];
#pragma unroll
    for (int r = 0; r < 8; ++r) {
      int m = m0 + wave * 16 + r + 8 * half;
      out[(size_t)m * 1024 + n] = acc[s][r] + bsv;
    }
  }
}

// ----------------------------------------------------------------- launch ----

extern "C" void kernel_launch(void* const* d_in, const int* in_sizes, int n_in,
                              void* d_out, int out_size, void* d_ws, size_t ws_size,
                              hipStream_t stream) {
  const float* x  = (const float*)d_in[0];
  const float* wq = (const float*)d_in[1];
  const float* bq = (const float*)d_in[2];
  const float* wk = (const float*)d_in[3];
  const float* bk = (const float*)d_in[4];
  const float* wv = (const float*)d_in[5];
  const float* bv = (const float*)d_in[6];
  const float* wo = (const float*)d_in[7];
  const float* bo = (const float*)d_in[8];

  const size_t NX = (size_t)2 * 2048 * 1024;  // 4,194,304
  const size_t NW = (size_t)1024 * 1024;      // 1,048,576

  char* ws = (char*)d_ws;
  u16* xb  = (u16*)(ws);                                   // 8 MB
  u16* wqt = (u16*)(ws + 8388608);                         // 2 MB each, [N][K]
  u16* wkt = (u16*)(ws + 10485760);
  u16* wvt = (u16*)(ws + 12582912);
  u16* wot = (u16*)(ws + 14680064);
  u16* Qb  = (u16*)(ws + 16777216);                        // 8 MB each
  u16* Kb  = (u16*)(ws + 25165824);
  u16* Vb  = (u16*)(ws + 33554432);                        // [B*H][64][2048]
  u16* Ob  = (u16*)(ws + 41943040);                        // ends at 48 MB

  cvt_f32_f16_kernel<<<(int)(NX / 256), 256, 0, stream>>>(x, xb, (int)NX);
  cvt_transpose_kernel<<<(int)(NW / 256), 256, 0, stream>>>(wq, wqt);
  cvt_transpose_kernel<<<(int)(NW / 256), 256, 0, stream>>>(wk, wkt);
  cvt_transpose_kernel<<<(int)(NW / 256), 256, 0, stream>>>(wv, wvt);
  cvt_transpose_kernel<<<(int)(NW / 256), 256, 0, stream>>>(wo, wot);

  dim3 gqkv(64, 16, 3);
  qkv_gemm_kernel<<<gqkv, 128, 0, stream>>>(xb, wqt, wkt, wvt, bq, bk, bv, Qb, Kb, Vb);

  dim3 gattn(32, 32);
  flash_attn_kernel<<<gattn, 128, 0, stream>>>(Qb, Kb, Vb, Ob);

  dim3 gproj(64, 16);
  out_proj_kernel<<<gproj, 128, 0, stream>>>(Ob, wot, bo, (float*)d_out);
}
